// Decoder3_21242908246276
// MI455X (gfx1250) — compile-verified, matrix operand order checked by reading, not run on hardware
//
#include <hip/hip_runtime.h>
#include <stdint.h>

// ---------------- problem constants ----------------
#define B_SZ   2048
#define N_NODE 129
#define H_DIM  128
#define N_HEAD 8
#define T_STEP 128
#define CLIPV  10.0f
#define NEGF   (-1e9f)
#define INV_SQRT_H 0.08838834764831845f   // 1/sqrt(128); 1/sqrt(DK)=0.25 folded into Q

typedef __attribute__((ext_vector_type(2)))  float    v2f;
typedef __attribute__((ext_vector_type(8)))  float    v8f;
typedef __attribute__((ext_vector_type(16))) _Float16 v16h;

// ---------------- LDS layout (bytes) ----------------
// A: embs[129][132] f32  -> per-wave 16x144 f32 score/compat slices
// B: per-wave 16x132 f32 (q_in -> Q)
// P: per-wave 16x160 f16 (attn probs) -> 16x132 f16 (G / Gout)
#define OFF_A     0
#define OFF_B     73728
#define OFF_P     141312
#define OFF_K     182272      // f32 K [129][132]
#define OFF_VT    250384      // f16 V^T [128][132]
#define OFF_K2    284176      // f16 K2 [129][132]
#define OFF_AID   318232      // int[128]
#define OFF_MASKW 318744      // uint[128*5]
#define OFF_GATES 321304      // float[128]
#define OFF_POOLP 321816      // float[128]
#define OFF_RS    322328      // float[128]
#define OFF_RED   322840      // float[24]
#define LDS_TOTAL 322944

__device__ __forceinline__ v8f wmma_k4(v2f a, v2f b, v8f c) {
  // D(16x16,f32) += A(16x4,f32) x B(4x16,f32)
  return __builtin_amdgcn_wmma_f32_16x16x4_f32(false, a, false, b, (short)0, c, false, false);
}
__device__ __forceinline__ v8f wmma_k32(v16h a, v16h b, v8f c) {
  // D(16x16,f32) += A(16x32,f16) x B(32x16,f16)
  return __builtin_amdgcn_wmma_f32_16x16x32_f16(false, a, false, b, (short)0, c, false, false);
}

union HPack   { unsigned u; _Float16 h[2]; };
union AFrag16 { v16h v; unsigned u[8]; };

// ---- gfx1250 async global->LDS copy (ASYNCcnt path), guarded so a missing
// ---- builtin falls back to plain loads without breaking the build.
#if __has_builtin(__builtin_amdgcn_global_load_async_to_lds_b128)
#define HAVE_ASYNC_LDS 1
typedef int v4i_ls __attribute__((__vector_size__(16)));
typedef __attribute__((address_space(1))) v4i_ls* gptr128;
typedef __attribute__((address_space(3))) v4i_ls* lptr128;
__device__ __forceinline__ void async_copy16(const void* g, void* l) {
  __builtin_amdgcn_global_load_async_to_lds_b128(
      (gptr128)(uintptr_t)g, (lptr128)(unsigned)(uintptr_t)l, 0, 0);
}
__device__ __forceinline__ void async_wait0() {
#if __has_builtin(__builtin_amdgcn_s_wait_asynccnt)
  __builtin_amdgcn_s_wait_asynccnt(0);
#else
  asm volatile("s_wait_asynccnt 0x0" ::: "memory");
#endif
}
#else
#define HAVE_ASYNC_LDS 0
#endif

__global__ void entropy_reduce_kernel(const float* __restrict__ ws, float* __restrict__ out) {
  __shared__ float s[256];
  float v = 0.0f;
  for (int i = threadIdx.x; i < B_SZ; i += 256) v += ws[i];   // fixed per-thread order
  s[threadIdx.x] = v;
  __syncthreads();
  for (int off = 128; off; off >>= 1) {
    if ((int)threadIdx.x < off) s[threadIdx.x] += s[threadIdx.x + off];
    __syncthreads();
  }
  if (threadIdx.x == 0) out[0] = s[0] * (1.0f / (float)B_SZ);
}

__global__ __launch_bounds__(256, 1)
void decoder_ptrnet_kernel(const float* __restrict__ embs, const float* __restrict__ pool,
                           const uint8_t* __restrict__ masks, const int* __restrict__ actions,
                           const float* __restrict__ Wfc, const float* __restrict__ Wfc1,
                           const float* __restrict__ Wq, const float* __restrict__ Wk,
                           const float* __restrict__ Wv, const float* __restrict__ Wo,
                           const float* __restrict__ Wkl, float* __restrict__ out,
                           float* __restrict__ entws)
{
  extern __shared__ char smem[];
  float*     E     = (float*)(smem + OFF_A);     // embs (phase 1)
  float*     Af    = (float*)(smem + OFF_A);     // per-wave score/compat slices
  float*     Bf    = (float*)(smem + OFF_B);
  _Float16*  Ph    = (_Float16*)(smem + OFF_P);
  float*     Kl    = (float*)(smem + OFF_K);
  _Float16*  Vt    = (_Float16*)(smem + OFF_VT);
  _Float16*  K2h   = (_Float16*)(smem + OFF_K2);
  int*       aid   = (int*)(smem + OFF_AID);
  unsigned*  mw    = (unsigned*)(smem + OFF_MASKW);
  float*     gates = (float*)(smem + OFF_GATES);
  float*     poolp = (float*)(smem + OFF_POOLP);
  float*     rsum  = (float*)(smem + OFF_RS);
  float*     red   = (float*)(smem + OFF_RED);

  const int b   = blockIdx.x;
  const int tid = threadIdx.x;
  const int w   = tid >> 5;        // wave 0..7
  const int lid = tid & 31;
  const int hf  = lid >> 4;        // lane half (0/1)
  const int l16 = lid & 15;

  // prime L2 with the weight matrices (64KB each; reused by every block)
  __builtin_prefetch(Wfc  + tid * 64, 0, 1);
  __builtin_prefetch(Wq   + tid * 64, 0, 1);
  __builtin_prefetch(Wk   + tid * 64, 0, 1);
  __builtin_prefetch(Wv   + tid * 64, 0, 1);
  __builtin_prefetch(Wo   + tid * 64, 0, 1);
  __builtin_prefetch(Wkl  + tid * 64, 0, 1);

  // ---------- load embs[b] -> LDS ----------
  const float* eb = embs + (size_t)b * N_NODE * H_DIM;
#if HAVE_ASYNC_LDS
  for (int i = tid; i < (N_NODE * H_DIM) / 4; i += 256) {   // 16B chunks
    const int r = i >> 5, c = (i & 31) * 4;
    async_copy16(eb + r * H_DIM + c, E + r * 132 + c);
  }
#else
  for (int i = tid; i < N_NODE * H_DIM; i += 256)
    E[(i >> 7) * 132 + (i & 127)] = eb[i];
#endif

  // aid[t] = previous action (aid[0] = depot 0)
  if (tid < T_STEP) aid[tid] = (tid == 0) ? 0 : actions[(tid - 1) * B_SZ + b];

  // pool_p = pool[b] @ Wfc1   (tiny 128x128 matvec)
  if (tid < H_DIM) {
    float s = 0.0f;
    for (int i = 0; i < H_DIM; ++i) s += pool[b * H_DIM + i] * Wfc1[i * H_DIM + tid];
    poolp[tid] = s;
  }

  // sequential mask walk (the only true recurrence; trivial bit ops)
  if (tid == 0) {
    unsigned wb[5] = {0, 0, 0, 0, 0};
    const uint8_t* mb = masks + (size_t)b * N_NODE;
    for (int n = 0; n < N_NODE; ++n) if (mb[n]) wb[n >> 5] |= (1u << (n & 31));
    bool dones = false;
    for (int t = 0; t < T_STEP; ++t) {
      mw[t * 5 + 0] = wb[0]; mw[t * 5 + 1] = wb[1]; mw[t * 5 + 2] = wb[2];
      mw[t * 5 + 3] = wb[3]; mw[t * 5 + 4] = wb[4];
      int a = actions[t * B_SZ + b];
      wb[a >> 5] &= ~(1u << (a & 31));
      unsigned any = wb[0] | wb[1] | wb[2] | wb[3] | wb[4];
      if (!any) { dones = true; wb[0] |= 1u; }     // depot reopens when exhausted
      gates[t] = dones ? 0.0f : 1.0f;
    }
  }
#if HAVE_ASYNC_LDS
  async_wait0();
#endif
  __syncthreads();

  // ---------- stage 1: K = E@Wk (f32), V^T = (E@Wv)^T (f16), K2 = E@Wkl (f16) ----------
  for (int job = w; job < 216; job += 8) {      // 3 mats * 9 n-tiles * 8 col-tiles
    const int m = job / 72, rem = job % 72;
    const int j = rem >> 3, c = rem & 7;
    const float* Wm = (m == 0) ? Wk : (m == 1) ? Wv : Wkl;
    int arow = j * 16 + l16; if (arow > 128) arow = 128;    // clamp padded tile
    const int col = c * 16 + l16;
    v8f acc = {};
    for (int k0 = 0; k0 < H_DIM; k0 += 4) {
      const int kk = k0 + 2 * hf;
      v2f a, bb;
      a.x  = E[arow * 132 + kk];       a.y  = E[arow * 132 + kk + 1];
      bb.x = Wm[kk * H_DIM + col];     bb.y = Wm[(kk + 1) * H_DIM + col];
      acc = wmma_k4(a, bb, acc);
    }
#pragma unroll
    for (int r = 0; r < 8; ++r) {
      const int row = j * 16 + r + 8 * hf;
      if (row < N_NODE) {
        if (m == 0)      Kl[row * 132 + col]  = acc[r];
        else if (m == 1) Vt[col * 132 + row]  = (_Float16)acc[r];   // transposed
        else             K2h[row * 132 + col] = (_Float16)acc[r];
      }
    }
  }
  __syncthreads();

  // each wave owns t-rows t0..t0+15
  const int t0 = w * 16;
  float*    Bw = Bf + w * 2112;            // 16 x 132 f32
  float*    Aw = Af + w * 2304;            // 16 x 144 f32
  _Float16* Pw = Ph + w * 2560;            // 16 x 160 f16
  _Float16* Gw = Ph + w * 2560;            // 16 x 132 f16 (same slice, reused)

  // ---------- q_in = X@Wfc + pool_p  (X rows gathered by aid) ----------
  const int xrow = aid[t0 + l16];
  v8f qa[8];
#pragma unroll
  for (int c = 0; c < 8; ++c) {
    v8f acc = {};
    const int col = c * 16 + l16;
    for (int k0 = 0; k0 < H_DIM; k0 += 4) {
      const int kk = k0 + 2 * hf;
      v2f a, bb;
      a.x  = E[xrow * 132 + kk];        a.y  = E[xrow * 132 + kk + 1];
      bb.x = Wfc[kk * H_DIM + col];     bb.y = Wfc[(kk + 1) * H_DIM + col];
      acc = wmma_k4(a, bb, acc);
    }
    const float pv = poolp[col];
#pragma unroll
    for (int r = 0; r < 8; ++r) acc[r] += pv;
    qa[c] = acc;
  }
  __syncthreads();                                   // all waves done reading E
#pragma unroll
  for (int c = 0; c < 8; ++c)
#pragma unroll
    for (int r = 0; r < 8; ++r)
      Bw[(r + 8 * hf) * 132 + c * 16 + l16] = qa[c][r];
  __syncthreads();

  // ---------- Q = (q_in @ Wq) * (1/sqrt(DK)) ----------
  v8f qq[8];
#pragma unroll
  for (int c = 0; c < 8; ++c) {
    v8f acc = {};
    const int col = c * 16 + l16;
    for (int k0 = 0; k0 < H_DIM; k0 += 4) {
      const int kk = k0 + 2 * hf;
      v2f a, bb;
      a.x  = Bw[l16 * 132 + kk];        a.y  = Bw[l16 * 132 + kk + 1];
      bb.x = Wq[kk * H_DIM + col];      bb.y = Wq[(kk + 1) * H_DIM + col];
      acc = wmma_k4(a, bb, acc);
    }
#pragma unroll
    for (int r = 0; r < 8; ++r) acc[r] *= 0.25f;
    qq[c] = acc;
  }
  __syncthreads();
#pragma unroll
  for (int c = 0; c < 8; ++c)
#pragma unroll
    for (int r = 0; r < 8; ++r)
      Bw[(r + 8 * hf) * 132 + c * 16 + l16] = qq[c][r];
  __syncthreads();

  // ---------- per-head attention ----------
  v8f gfr[8];
#pragma unroll
  for (int h = 0; h < N_HEAD; ++h) {
    // scores = Q_h @ K_h^T  (f32 WMMA, K-dim = 16)
    for (int j = 0; j < 9; ++j) {
      int nrow = j * 16 + l16; if (nrow > 128) nrow = 128;
      v8f sc = {};
#pragma unroll
      for (int k0 = 0; k0 < 16; k0 += 4) {
        const int kk = h * 16 + k0 + 2 * hf;
        v2f a, bb;
        a.x  = Bw[l16 * 132 + kk];       a.y  = Bw[l16 * 132 + kk + 1];
        bb.x = Kl[nrow * 132 + kk];      bb.y = Kl[nrow * 132 + kk + 1];
        sc = wmma_k4(a, bb, sc);
      }
#pragma unroll
      for (int r = 0; r < 8; ++r)
        Aw[(r + 8 * hf) * 144 + j * 16 + l16] = sc[r];
    }
    // masked softmax over nodes; two lanes per t-row; probs -> f16
    {
      const int row = l16, t = t0 + row;
      const unsigned* mwt = &mw[t * 5];
      float mx = NEGF;
      for (int n = hf; n < N_NODE; n += 2)
        if ((mwt[n >> 5] >> (n & 31)) & 1u)
          mx = fmaxf(mx, Aw[row * 144 + n]);
      mx = fmaxf(mx, __shfl_xor(mx, 16, 32));
      float sum = 0.0f;
      for (int n = hf; n <= 128; n += 2) {
        float p = 0.0f;
        if ((mwt[n >> 5] >> (n & 31)) & 1u)
          p = __expf(Aw[row * 144 + n] - mx);
        Pw[row * 160 + n] = (_Float16)p;
        sum += p;
      }
      sum += __shfl_xor(sum, 16, 32);
      if (hf == 0) rsum[t] = sum;
    }
    // glimpse_h = P @ V_h  (f16 WMMA, K=128) + rank-1 update for node 128
    v8f gl = {};
    for (int k0 = 0; k0 < 128; k0 += 32) {
      AFrag16 a, bb;
#pragma unroll
      for (int r = 0; r < 4; ++r) {
        const int ka = k0 + 2 * r + 8 * hf;
        a.u[r]      = *(const unsigned*)&Pw[l16 * 160 + ka];
        a.u[4 + r]  = *(const unsigned*)&Pw[l16 * 160 + ka + 16];
        bb.u[r]     = *(const unsigned*)&Vt[(h * 16 + l16) * 132 + ka];
        bb.u[4 + r] = *(const unsigned*)&Vt[(h * 16 + l16) * 132 + ka + 16];
      }
      gl = wmma_k32(a.v, bb.v, gl);
    }
    {
      const float v128 = (float)Vt[(h * 16 + l16) * 132 + 128];
#pragma unroll
      for (int r = 0; r < 8; ++r) {
        const int trow = t0 + r + 8 * hf;
        gl[r] += (float)Pw[(r + 8 * hf) * 160 + 128] * v128;
        gl[r] /= rsum[trow];
      }
    }
    gfr[h] = gl;
  }
  __syncthreads();

  // ---------- store G (f16), Gout = G@Wo, compat = Gout@K2^T / sqrt(H) ----------
#pragma unroll
  for (int h = 0; h < 8; ++h)
#pragma unroll
    for (int r = 0; r < 8; ++r)
      Gw[(r + 8 * hf) * 132 + h * 16 + l16] = (_Float16)gfr[h][r];

  v8f go[8];
#pragma unroll
  for (int c = 0; c < 8; ++c) {
    v8f acc = {};
    const int col = c * 16 + l16;
    for (int k0 = 0; k0 < 128; k0 += 32) {
      AFrag16 a, bb;
#pragma unroll
      for (int r = 0; r < 4; ++r) {
        const int ka = k0 + 2 * r + 8 * hf;
        a.u[r]     = *(const unsigned*)&Gw[l16 * 132 + ka];
        a.u[4 + r] = *(const unsigned*)&Gw[l16 * 132 + ka + 16];
        HPack p0, p1;
        p0.h[0] = (_Float16)Wo[ka * H_DIM + col];
        p0.h[1] = (_Float16)Wo[(ka + 1) * H_DIM + col];
        p1.h[0] = (_Float16)Wo[(ka + 16) * H_DIM + col];
        p1.h[1] = (_Float16)Wo[(ka + 17) * H_DIM + col];
        bb.u[r] = p0.u; bb.u[4 + r] = p1.u;
      }
      acc = wmma_k32(a.v, bb.v, acc);
    }
    go[c] = acc;
  }
#pragma unroll
  for (int c = 0; c < 8; ++c)
#pragma unroll
    for (int r = 0; r < 8; ++r)
      Gw[(r + 8 * hf) * 132 + c * 16 + l16] = (_Float16)go[c][r];

  for (int j = 0; j < 9; ++j) {
    int nrow = j * 16 + l16; if (nrow > 128) nrow = 128;
    v8f acc = {};
    for (int k0 = 0; k0 < 128; k0 += 32) {
      AFrag16 a, bb;
#pragma unroll
      for (int r = 0; r < 4; ++r) {
        const int ka = k0 + 2 * r + 8 * hf;
        a.u[r]      = *(const unsigned*)&Gw[l16 * 132 + ka];
        a.u[4 + r]  = *(const unsigned*)&Gw[l16 * 132 + ka + 16];
        bb.u[r]     = *(const unsigned*)&K2h[nrow * 132 + ka];
        bb.u[4 + r] = *(const unsigned*)&K2h[nrow * 132 + ka + 16];
      }
      acc = wmma_k32(a.v, bb.v, acc);
    }
#pragma unroll
    for (int r = 0; r < 8; ++r)
      Aw[(r + 8 * hf) * 144 + j * 16 + l16] = acc[r] * INV_SQRT_H;
  }

  // ---------- logits / log-softmax / lp / entropy per t-row ----------
  float lp = 0.0f, ent = 0.0f, gt = 0.0f;
  {
    const int row = l16, t = t0 + row;
    const unsigned* mwt = &mw[t * 5];
    float mx = NEGF;
    for (int n = hf; n < N_NODE; n += 2)
      if ((mwt[n >> 5] >> (n & 31)) & 1u)
        mx = fmaxf(mx, CLIPV * tanhf(Aw[row * 144 + n]));
    mx = fmaxf(mx, __shfl_xor(mx, 16, 32));
    float sa = 0.0f, sb = 0.0f;
    for (int n = hf; n < N_NODE; n += 2)
      if ((mwt[n >> 5] >> (n & 31)) & 1u) {
        const float l = CLIPV * tanhf(Aw[row * 144 + n]);
        const float e = __expf(l - mx);
        sa += e; sb += e * l;
      }
    sa += __shfl_xor(sa, 16, 32);
    sb += __shfl_xor(sb, 16, 32);
    const float lse = mx + __logf(sa);
    const int   act = actions[t * B_SZ + b];
    const float la  = CLIPV * tanhf(Aw[row * 144 + act]);
    const float g   = gates[t];
    lp  = (la - lse) * g;
    ent = (lse - sb / sa) * g;
    gt  = g;
    if (hf) { lp = 0.0f; ent = 0.0f; gt = 0.0f; }   // count each row once
  }
#pragma unroll
  for (int off = 16; off; off >>= 1) {
    lp  += __shfl_xor(lp, off, 32);
    ent += __shfl_xor(ent, off, 32);
    gt  += __shfl_xor(gt, off, 32);
  }
  if (lid == 0) { red[w * 3] = lp; red[w * 3 + 1] = ent; red[w * 3 + 2] = gt; }
  __syncthreads();
  if (tid == 0) {
    float slp = 0.0f, sent = 0.0f, sg = 0.0f;
    for (int i = 0; i < 8; ++i) { slp += red[i * 3]; sent += red[i * 3 + 1]; sg += red[i * 3 + 2]; }
    out[1 + b] = slp / sg;       // log_probs[b]
    entws[b]   = sent;           // per-block entropy partial (reduced deterministically)
  }
}

extern "C" void kernel_launch(void* const* d_in, const int* in_sizes, int n_in,
                              void* d_out, int out_size, void* d_ws, size_t ws_size,
                              hipStream_t stream) {
  (void)in_sizes; (void)n_in; (void)out_size; (void)ws_size;
  const float*   embs    = (const float*)d_in[0];
  const float*   pool    = (const float*)d_in[1];
  const uint8_t* masks   = (const uint8_t*)d_in[2];
  const int*     actions = (const int*)d_in[3];
  const float*   Wfc     = (const float*)d_in[4];
  const float*   Wfc1    = (const float*)d_in[5];
  const float*   Wq      = (const float*)d_in[6];
  const float*   Wk      = (const float*)d_in[7];
  const float*   Wv      = (const float*)d_in[8];
  const float*   Wo      = (const float*)d_in[9];
  const float*   Wkl     = (const float*)d_in[10];
  float*         out     = (float*)d_out;
  float*         entws   = (float*)d_ws;

  (void)hipFuncSetAttribute((const void*)decoder_ptrnet_kernel,
                            hipFuncAttributeMaxDynamicSharedMemorySize, LDS_TOTAL);
  hipLaunchKernelGGL(decoder_ptrnet_kernel, dim3(B_SZ), dim3(256), LDS_TOTAL, stream,
                     embs, pool, masks, actions, Wfc, Wfc1, Wq, Wk, Wv, Wo, Wkl, out, entws);
  hipLaunchKernelGGL(entropy_reduce_kernel, dim3(1), dim3(256), 0, stream, entws, out);
}
